// AutoRoundGPTQLinear_77292231459025
// MI455X (gfx1250) — matrix operand order repaired and checked by reading, not compiled, not run on hardware
//
#include <hip/hip_runtime.h>

typedef __attribute__((ext_vector_type(16))) _Float16 v16h;
typedef __attribute__((ext_vector_type(2)))  _Float16 h2;
typedef __attribute__((ext_vector_type(8)))  float    v8f;
typedef __attribute__((ext_vector_type(4)))  float    f32x4;

#define IN_F    4096
#define OUT_F   11008
#define GSIZE   128
#define BM      256
#define BN      256
#define BK      32
#define LDK     40                 // padded K stride in halves (80B: 16B aligned, bank-spread)
#define KTILES  (IN_F / BK)        // 128
#define M_TOTAL 8192               // 4 * 2048
#define NTHREADS 512

union AFrag  { v16h h; f32x4 f[2]; };
union BStage { h2 p[8]; f32x4 f[2]; };
union HBits  { unsigned u; h2 h; };

__global__ __launch_bounds__(NTHREADS)
void gptq_wmma_kernel(const float* __restrict__ x,
                      const int*   __restrict__ qweight,
                      const int*   __restrict__ qzeros,
                      const _Float16* __restrict__ scales,
                      float* __restrict__ out)
{
    __shared__ __align__(16) _Float16 As[2][BM][LDK];
    __shared__ __align__(16) _Float16 Bs[2][BN][LDK];

    const int tid  = threadIdx.x;
    const int wid  = tid >> 5;      // 0..15
    const int lane = tid & 31;
    const int half = lane >> 4;     // 16-lane group
    const int lrow = lane & 15;

    const int m0 = blockIdx.y * BM;
    const int n0 = blockIdx.x * BN;

    const int wm = (wid & 3) * 64;  // wave sub-tile: 64 rows
    const int wn = (wid >> 2) * 64; //               64 cols

    // B-fill mapping: one column per thread, two packed words (2 of 4 k-rows)
    const int bn_   = tid & 255;
    const int krow0 = (tid >> 8) * 2;        // 0 or 2
    const int gncol = n0 + bn_;

    v8f acc[4][4];
    #pragma unroll
    for (int mi = 0; mi < 4; ++mi)
        #pragma unroll
        for (int ni = 0; ni < 4; ++ni)
            #pragma unroll
            for (int r = 0; r < 8; ++r)
                acc[mi][ni][r] = 0.0f;

    // ---- global prefetch (registers) ----
    float4 ar[4];
    int qr0, qr1;
    h2 s2, c2;                      // splat(scale), splat(-(1024+zero)*scale)

    auto loadGlobal = [&](int kb) {
        #pragma unroll
        for (int i = 0; i < 4; ++i) {
            int idx = tid + i * NTHREADS;    // 0..2047
            int row = idx >> 3;
            int c4  = idx & 7;
            ar[i] = *reinterpret_cast<const float4*>(
                x + (size_t)(m0 + row) * IN_F + kb + c4 * 4);
        }
        qr0 = qweight[(size_t)((kb >> 3) + krow0)     * OUT_F + gncol];
        qr1 = qweight[(size_t)((kb >> 3) + krow0 + 1) * OUT_F + gncol];
        if ((kb & (GSIZE - 1)) == 0) {       // uniform: group changes every 4th tile
            int g  = kb >> 7;
            int zq = qzeros[g * (OUT_F / 8) + (gncol >> 3)];
            float zf = (float)((zq >> ((gncol & 7) * 4)) & 15);
            float sf = (float)scales[g * OUT_F + gncol];
            float cf = -(1024.0f + zf) * sf;
            s2[0] = (_Float16)sf; s2[1] = (_Float16)sf;
            c2[0] = (_Float16)cf; c2[1] = (_Float16)cf;
        }
    };

    auto storeLds = [&](int buf) {
        #pragma unroll
        for (int i = 0; i < 4; ++i) {
            int idx = tid + i * NTHREADS;
            int row = idx >> 3;
            int c4  = idx & 7;
            _Float16* p = &As[buf][row][c4 * 4];
            p[0] = (_Float16)ar[i].x;
            p[1] = (_Float16)ar[i].y;
            p[2] = (_Float16)ar[i].z;
            p[3] = (_Float16)ar[i].w;
        }
        // int4 -> fp16 via exponent trick: fp16(0x6400|n) == 1024+n (exact),
        // (n - z)*s == (1024+n)*s - (1024+z)*s  -> one v_pk_fma_f16 per pair
        BStage st;
        #pragma unroll
        for (int j = 0; j < 4; ++j) {
            HBits a, b;
            unsigned u0 = ((unsigned)qr0 >> (8 * j)) & 0xFFu;
            unsigned u1 = ((unsigned)qr1 >> (8 * j)) & 0xFFu;
            a.u = 0x64006400u | (u0 & 0xFu) | ((u0 & 0xF0u) << 12);
            b.u = 0x64006400u | (u1 & 0xFu) | ((u1 & 0xF0u) << 12);
            st.p[j]     = a.h * s2 + c2;
            st.p[4 + j] = b.h * s2 + c2;
        }
        f32x4* dst = reinterpret_cast<f32x4*>(&Bs[buf][bn_][krow0 * 8]);
        dst[0] = st.f[0];
        dst[1] = st.f[1];
    };

    auto compute = [&](int buf) {
        AFrag a[4], b[4];
        // A 16x32 f16 frag: lanes0-15 K0-7|16-23, lanes16-31 K8-15|24-31
        #pragma unroll
        for (int mi = 0; mi < 4; ++mi) {
            const _Float16* ap = &As[buf][wm + mi * 16 + lrow][half * 8];
            a[mi].f[0] = *reinterpret_cast<const f32x4*>(ap);
            a[mi].f[1] = *reinterpret_cast<const f32x4*>(ap + 16);
        }
        // B 32x16 f16 frag: lanes0-15 K0-7|16-23, lanes16-31 K8-15|24-31 (N-major LDS)
        #pragma unroll
        for (int ni = 0; ni < 4; ++ni) {
            const _Float16* bp = &Bs[buf][wn + ni * 16 + lrow][half * 8];
            b[ni].f[0] = *reinterpret_cast<const f32x4*>(bp);
            b[ni].f[1] = *reinterpret_cast<const f32x4*>(bp + 16);
        }
        #pragma unroll
        for (int mi = 0; mi < 4; ++mi)
            #pragma unroll
            for (int ni = 0; ni < 4; ++ni)
                acc[mi][ni] = __builtin_amdgcn_wmma_f32_16x16x32_f16(
                    false, a[mi].h, false, b[ni].h,
                    (short)0, acc[mi][ni], false, false);
    };

    // ---- pipeline: LDS double buffer, one barrier per K-tile ----
    loadGlobal(0);
    storeLds(0);
    __syncthreads();

    for (int kt = 0; kt < KTILES; ++kt) {
        const int cur = kt & 1;
        const bool more = (kt + 1) < KTILES;
        if (more) loadGlobal((kt + 1) * BK);
        compute(cur);
        if (more) storeLds(cur ^ 1);
        __syncthreads();
    }

    // ---- epilogue: C/D layout -> fp32 output ----
    const int rbase = m0 + wm + half * 8;
    const int cbase = n0 + wn + lrow;
    #pragma unroll
    for (int mi = 0; mi < 4; ++mi)
        #pragma unroll
        for (int ni = 0; ni < 4; ++ni) {
            float* op = out + (size_t)(rbase + mi * 16) * OUT_F + cbase + ni * 16;
            #pragma unroll
            for (int r = 0; r < 8; ++r)
                op[(size_t)r * OUT_F] = acc[mi][ni][r];
        }
}

extern "C" void kernel_launch(void* const* d_in, const int* in_sizes, int n_in,
                              void* d_out, int out_size, void* d_ws, size_t ws_size,
                              hipStream_t stream) {
    const float*    x  = (const float*)d_in[0];
    const int*      qw = (const int*)d_in[1];
    const int*      qz = (const int*)d_in[2];
    const _Float16* sc = (const _Float16*)d_in[3];
    float*          o  = (float*)d_out;

    dim3 grid(OUT_F / BN, M_TOTAL / BM);   // (43, 32)
    gptq_wmma_kernel<<<grid, NTHREADS, 0, stream>>>(x, qw, qz, sc, o);
}
